// HungarianMatcher_36962488549796
// MI455X (gfx1250) — matrix operand order repaired and checked by reading, not compiled, not run on hardware
//
#include <hip/hip_runtime.h>

typedef __attribute__((ext_vector_type(16))) _Float16 v16h;
typedef __attribute__((ext_vector_type(8)))  _Float16 v8h;
typedef __attribute__((ext_vector_type(8)))  float    v8f;
typedef __attribute__((ext_vector_type(4)))  float    v4f;
typedef __attribute__((ext_vector_type(2)))  float    v2f;

#define BS    32
#define NQ    1000
#define NCLS  100
#define KPAD  128
#define TPS   50
#define TT    (BS * TPS)      // 1600 total targets
#define NROWS (BS * NQ)       // 32000 query rows
#define NTT   (TT / 16)       // 100 target tiles
#define NMT   (NROWS / 16)    // 2000 query tiles
#define NT    256             // LSA block size

// ---------------------------------------------------------------------------
// Kernel 1: focal-loss class terms, padded to KPAD classes, stored f16.
// ---------------------------------------------------------------------------
__global__ void posneg_kernel(const float* __restrict__ logits,
                              _Float16* __restrict__ posneg) {
    int idx = blockIdx.x * blockDim.x + threadIdx.x;   // NROWS*KPAD threads
    if (idx >= NROWS * KPAD) return;
    int q = idx >> 7;
    int c = idx & (KPAD - 1);
    float r = 0.0f;
    if (c < NCLS) {
        float x = logits[q * NCLS + c];
        float p = 1.0f / (1.0f + __expf(-x));
        float pos = 0.25f * (1.0f - p) * (1.0f - p) * (-__logf(p + 1e-8f));
        float neg = 0.75f * p * p * (-__logf(1.0f - p + 1e-8f));
        r = pos - neg;
    }
    posneg[idx] = (_Float16)r;
}

// ---------------------------------------------------------------------------
// Kernel 2a: pre-swizzled one-hot B tiles, in cost_kernel's per-lane B order (k = 16*half + s).
// bswz[((nt*4 + kc)*32 + lane)*16 + h] = (tgt_ids[nt*16 + (lane&15)] ==
//                                         kc*32 + (lane>>4)*16 + h)
// 400 KB total; L2-resident, reused by all 2000 query tiles.
// ---------------------------------------------------------------------------
__global__ void btab_kernel(const int* __restrict__ tgt_ids,
                            _Float16* __restrict__ bswz) {
    int tid = blockIdx.x * blockDim.x + threadIdx.x;   // NTT*4*32 = 12800
    if (tid >= NTT * 4 * 32) return;
    int nt   = tid / (4 * 32);
    int kc   = (tid / 32) & 3;
    int lane = tid & 31;
    int g    = lane >> 4;
    int n16  = lane & 15;
    int tcls = tgt_ids[nt * 16 + n16];
    int k0   = kc * 32 + g * 16;
    v8h lo, hi;
#pragma unroll
    for (int h = 0; h < 8; ++h) {
        lo[h] = (tcls == (k0 + h))     ? (_Float16)1.0f : (_Float16)0.0f;
        hi[h] = (tcls == (k0 + 8 + h)) ? (_Float16)1.0f : (_Float16)0.0f;
    }
    _Float16* dst = bswz + (size_t)tid * 16;
    *(v8h*)dst       = lo;
    *(v8h*)(dst + 8) = hi;
}

// ---------------------------------------------------------------------------
// Kernel 2b/2c: per-query / per-target geometry tables {c, w, s, e}.
// ---------------------------------------------------------------------------
__global__ void geom_kernel(const float* __restrict__ boxes,
                            v4f* __restrict__ tab, int n) {
    int i = blockIdx.x * blockDim.x + threadIdx.x;
    if (i >= n) return;
    v2f b = *(const v2f*)(boxes + 2 * i);
    v4f t;
    t.x = b.x;                                               // center
    t.y = b.y;                                               // width
    t.z = fminf(fmaxf(b.x - 0.5f * b.y, 0.0f), 1.0f);        // s
    t.w = fminf(fmaxf(b.x + 0.5f * b.y, 0.0f), 1.0f);        // e
    tab[i] = t;
}

// ---------------------------------------------------------------------------
// Kernel 3 (WMMA): one 16x16 tile of C per wave.
//   class cost via 4x wmma_f32_16x16x32_f16 with cached pre-swizzled one-hot B.
//   bbox L1 + IoU from precomputed geometry tables, fused per C/D-layout lane.
// ---------------------------------------------------------------------------
__global__ void cost_kernel(const _Float16* __restrict__ posneg,
                            const _Float16* __restrict__ bswz,
                            const v4f* __restrict__ qtab,
                            const v4f* __restrict__ ttab,
                            float* __restrict__ C) {
    int wave = (blockIdx.x * blockDim.x + threadIdx.x) >> 5;
    int lane = threadIdx.x & 31;
    int mt = wave / NTT;                // 0..1999  (query tile)
    int nt = wave % NTT;                // 0..99    (target tile)
    int g   = lane >> 4;                // half-wave group
    int n16 = lane & 15;

    const _Float16* arow  = posneg + (size_t)(mt * 16 + n16) * KPAD;
    const _Float16* bbase = bswz + (size_t)nt * 2048 + lane * 16;
    int tcol = nt * 16 + n16;

    v8f acc = {};
#pragma unroll
    for (int kc = 0; kc < 4; ++kc) {
        int k0 = kc * 32;
        v8h alo = *(const v8h*)(arow + k0 + g * 16);       // K = k0+g*16  .. +7
        v8h ahi = *(const v8h*)(arow + k0 + g * 16 + 8);   // K = k0+g*16+8.. +15
        v8h blo = *(const v8h*)(bbase + kc * 512);
        v8h bhi = *(const v8h*)(bbase + kc * 512 + 8);
        v16h a, b;
#pragma unroll
        for (int h = 0; h < 8; ++h) {
            a[h] = alo[h]; a[8 + h] = ahi[h];
            b[h] = blo[h]; b[8 + h] = bhi[h];
        }
        acc = __builtin_amdgcn_wmma_f32_16x16x32_f16(
            false, a, false, b, (short)0, acc, false, false);
    }

    v4f tt = ttab[tcol];                 // {tc, tw, s2, e2}
    float len2 = tt.w - tt.z;
#pragma unroll
    for (int r = 0; r < 8; ++r) {
        int qg = mt * 16 + r + 8 * g;    // C/D layout row
        v4f qq = qtab[qg];               // {qc, qw, s1, e1}
        float l1    = fabsf(qq.x - tt.x) + fabsf(qq.y - tt.y);
        float inter = fmaxf(fminf(qq.w, tt.w) - fmaxf(qq.z, tt.z), 0.0f);
        float uni   = (qq.w - qq.z) + len2 - inter;
        C[(size_t)qg * TT + tcol] = acc[r] + l1 - inter / uni;
    }
}

// ---------------------------------------------------------------------------
// Kernel 4: extract per-batch transposed submatrix Csub[b][i][j] = C[b,j,b*50+i]
// ---------------------------------------------------------------------------
__global__ void extract_kernel(const float* __restrict__ C,
                               float* __restrict__ Csub) {
    int idx = blockIdx.x * blockDim.x + threadIdx.x;    // BS*NQ*TPS
    if (idx >= BS * NQ * TPS) return;
    int b   = idx / (NQ * TPS);
    int rem = idx % (NQ * TPS);
    int j   = rem / TPS;          // query (column of LSA)
    int i   = rem % TPS;          // target (row of LSA)
    Csub[((size_t)b * TPS + i) * NQ + j] =
        C[((size_t)b * NQ + j) * TT + b * TPS + i];
}

// ---------------------------------------------------------------------------
// Kernel 5: Jonker-Volgenant shortest augmenting path, one workgroup per batch.
// ---------------------------------------------------------------------------
__global__ void lsa_kernel(const float* __restrict__ Csub,
                           float* __restrict__ out_idx) {
    __shared__ float v_[NQ + 1];
    __shared__ float minv[NQ + 1];
    __shared__ int   way[NQ + 1];
    __shared__ int   p_[NQ + 1];
    __shared__ unsigned char used[NQ + 1];
    __shared__ float u_[TPS + 1];
    __shared__ float redv[NT];
    __shared__ int   redj[NT];
    __shared__ int   j0s, j1s, i0s;
    __shared__ float deltas;

    int b = blockIdx.x;
    int tid = threadIdx.x;
    const float* cost = Csub + (size_t)b * TPS * NQ;

    for (int j = tid; j <= NQ; j += NT) { v_[j] = 0.0f; p_[j] = 0; }
    if (tid <= TPS) u_[tid] = 0.0f;
    __syncthreads();

    for (int i = 1; i <= TPS; ++i) {
        if (tid == 0) { p_[0] = i; j0s = 0; }
        for (int j = tid; j <= NQ; j += NT) { minv[j] = 1e30f; used[j] = 0; }
        __syncthreads();
        while (true) {
            if (tid == 0) { used[j0s] = 1; i0s = p_[j0s]; }
            __syncthreads();
            int i0 = i0s, jj0 = j0s;
            float ui0 = u_[i0];
            const float* crow = cost + (size_t)(i0 - 1) * NQ;
            for (int j = tid + 1; j <= NQ; j += NT) {
                if (!used[j]) {
                    float cur = crow[j - 1] - ui0 - v_[j];
                    if (cur < minv[j]) { minv[j] = cur; way[j] = jj0; }
                }
            }
            __syncthreads();
            float bv = 1e37f; int bj = 1;
            for (int j = tid + 1; j <= NQ; j += NT)
                if (!used[j] && minv[j] < bv) { bv = minv[j]; bj = j; }
            redv[tid] = bv; redj[tid] = bj;
            __syncthreads();
            for (int s = NT >> 1; s > 0; s >>= 1) {
                if (tid < s && redv[tid + s] < redv[tid]) {
                    redv[tid] = redv[tid + s]; redj[tid] = redj[tid + s];
                }
                __syncthreads();
            }
            if (tid == 0) { j1s = redj[0]; deltas = redv[0]; }
            __syncthreads();
            float delta = deltas;
            for (int j = tid; j <= NQ; j += NT) {
                if (used[j]) { u_[p_[j]] += delta; v_[j] -= delta; }
                else         { minv[j] -= delta; }
            }
            __syncthreads();
            if (tid == 0) j0s = j1s;
            __syncthreads();
            if (p_[j0s] == 0) break;
        }
        if (tid == 0) {   // augment along the path
            int j0 = j0s;
            while (j0 != 0) { int jn = way[j0]; p_[j0] = p_[jn]; j0 = jn; }
        }
        __syncthreads();
    }
    if (tid == 0) {       // emit [2, 50]: sorted query indices + matched targets
        int k = 0;
        float* ob = out_idx + b * 2 * TPS;
        for (int j = 1; j <= NQ && k < TPS; ++j) {
            if (p_[j] != 0) {
                ob[k] = (float)(j - 1);
                ob[TPS + k] = (float)(p_[j] - 1);
                ++k;
            }
        }
    }
}

// ---------------------------------------------------------------------------
extern "C" void kernel_launch(void* const* d_in, const int* in_sizes, int n_in,
                              void* d_out, int out_size, void* d_ws, size_t ws_size,
                              hipStream_t stream) {
    const float* logits     = (const float*)d_in[0];   // [32,1000,100]
    const float* pred_boxes = (const float*)d_in[1];   // [32,1000,2]
    const int*   tgt_ids    = (const int*)d_in[2];     // [1600]
    const float* tgt_bbox   = (const float*)d_in[3];   // [1600,2]

    float* C       = (float*)d_out;                       // [32,1000,1600]
    float* out_idx = C + (size_t)BS * NQ * TT;            // [32,2,50]

    // workspace carve-up
    char* ws = (char*)d_ws;
    _Float16* posneg = (_Float16*)ws;                               // 8.192 MB
    ws += (size_t)NROWS * KPAD * sizeof(_Float16);
    float* Csub = (float*)ws;                                       // 6.400 MB
    ws += (size_t)BS * TPS * NQ * sizeof(float);
    _Float16* bswz = (_Float16*)ws;                                 // 0.410 MB
    ws += (size_t)NTT * 4 * 32 * 16 * sizeof(_Float16);
    v4f* qtab = (v4f*)ws;                                           // 0.512 MB
    ws += (size_t)NROWS * sizeof(v4f);
    v4f* ttab = (v4f*)ws;                                           // 0.026 MB

    posneg_kernel<<<(NROWS * KPAD + 255) / 256, 256, 0, stream>>>(logits, posneg);
    btab_kernel<<<(NTT * 4 * 32 + 255) / 256, 256, 0, stream>>>(tgt_ids, bswz);
    geom_kernel<<<(NROWS + 255) / 256, 256, 0, stream>>>(pred_boxes, qtab, NROWS);
    geom_kernel<<<(TT + 255) / 256, 256, 0, stream>>>(tgt_bbox, ttab, TT);

    int waves = NMT * NTT;                     // 200000 tiles
    cost_kernel<<<waves / 8, 256, 0, stream>>>(posneg, bswz, qtab, ttab, C);

    extract_kernel<<<(BS * NQ * TPS + 255) / 256, 256, 0, stream>>>(C, Csub);

    lsa_kernel<<<BS, NT, 0, stream>>>(Csub, out_idx);
}
